// QNADE_43473658970605
// MI455X (gfx1250) — matrix operand order repaired and checked by reading, not compiled, not run on hardware
//
#include <hip/hip_runtime.h>
#include <hip/hip_bf16.h>
#include <hip/hip_fp16.h>

// ---------------------------------------------------------------------------
// QNADE on MI455X (gfx1250, wave32)
//   Phase 1: autoregressive sampling, one wave32 per sample chain,
//            incremental prefix update (latency-bound, VALU+LDS).
//   Phase 2: per-sample gradients via v_wmma_f32_16x16x32_f16
//            (xin is {-1,0,+1} -> exact in fp16; f32 accumulate),
//            W1 staged into LDS via GLOBAL_LOAD_ASYNC_TO_LDS_B128,
//            A-fragments hoisted and reused across tiles.
// ---------------------------------------------------------------------------

#define D_IN  64
#define H_DIM 256
#define NS    2048
#define SPB   8      // samples per block in the gradient kernel

typedef _Float16 v16h __attribute__((ext_vector_type(16)));
typedef _Float16 v8h  __attribute__((ext_vector_type(8)));
typedef float    v8f  __attribute__((ext_vector_type(8)));
typedef int      v4i  __attribute__((vector_size(16)));   // matches builtin param

#if __has_builtin(__builtin_amdgcn_global_load_async_to_lds_b128)
#define HAVE_ASYNC_LDS 1
#endif

__device__ __forceinline__ unsigned lane_id() {
  // wave32: mbcnt_lo over a full mask == lane index
  return __builtin_amdgcn_mbcnt_lo(~0u, 0u);
}

// gfx1250 has V_TANH_F32; use it if the builtin is exposed, else libm.
__device__ __forceinline__ float fast_tanh(float x) {
#if __has_builtin(__builtin_amdgcn_tanhf)
  return __builtin_amdgcn_tanhf(x);
#else
  return tanhf(x);
#endif
}

__device__ __forceinline__ unsigned rotl32(unsigned x, int r) {
  return (x << r) | (x >> (32 - r));
}

// Threefry-2x32-20 counter-based PRNG (key = (1, d), ctr = (sample, 0)).
__device__ __forceinline__ uint2 threefry2x32(unsigned k0, unsigned k1,
                                              unsigned c0, unsigned c1) {
  const unsigned ks2 = k0 ^ k1 ^ 0x1BD11BDAu;
  unsigned x0 = c0 + k0, x1 = c1 + k1;
  const int RA[4] = {13, 15, 26, 6};
  const int RB[4] = {17, 29, 16, 24};
#pragma unroll
  for (int i = 0; i < 4; i++) { x0 += x1; x1 = rotl32(x1, RA[i]); x1 ^= x0; }
  x0 += k1;  x1 += ks2 + 1u;
#pragma unroll
  for (int i = 0; i < 4; i++) { x0 += x1; x1 = rotl32(x1, RB[i]); x1 ^= x0; }
  x0 += ks2; x1 += k0 + 2u;
#pragma unroll
  for (int i = 0; i < 4; i++) { x0 += x1; x1 = rotl32(x1, RA[i]); x1 ^= x0; }
  x0 += k0;  x1 += k1 + 3u;
#pragma unroll
  for (int i = 0; i < 4; i++) { x0 += x1; x1 = rotl32(x1, RB[i]); x1 ^= x0; }
  x0 += k1;  x1 += ks2 + 4u;
#pragma unroll
  for (int i = 0; i < 4; i++) { x0 += x1; x1 = rotl32(x1, RA[i]); x1 ^= x0; }
  x0 += ks2; x1 += k0 + 5u;
  return make_uint2(x0, x1);
}

// ---------------------------------------------------------------------------
// WMMA fragment loaders (contiguous-k layouts, 16-bit A 16x32 / B 32x16)
// ---------------------------------------------------------------------------

// A: row-major [M][K] tile. lane L: M = L&15; grp = L>>4;
// halves k = grp*16 + {0..7}  and  grp*16 + 8 + {0..7}  (two b128 LDS loads).
__device__ __forceinline__ v16h load_frag_a(const _Float16* __restrict__ base,
                                            int lda, unsigned lane) {
  const int m   = lane & 15;
  const int grp = (lane >> 4) & 1;
  const _Float16* row = base + m * lda;
  v8h lo = *(const v8h*)(row + grp * 16);
  v8h hi = *(const v8h*)(row + grp * 16 + 8);
  v16h a;
#pragma unroll
  for (int i = 0; i < 8; i++) { a[i] = lo[i]; a[i + 8] = hi[i]; }
  return a;
}

// B: stored transposed as [N][K]. lane L: N = L&15; grp = L>>4;
// halves k = grp*16 + {0..15}  (one contiguous 32B LDS load).
__device__ __forceinline__ v16h load_frag_b(const _Float16* __restrict__ baseT,
                                            int ldb, unsigned lane) {
  const int n   = lane & 15;
  const int grp = (lane >> 4) & 1;
  return *(const v16h*)(baseT + n * ldb + grp * 16);
}

// ---------------------------------------------------------------------------
// Phase 1: sampling.  One wave32 per sample; 8 samples (waves) per block.
// ---------------------------------------------------------------------------
__global__ __launch_bounds__(256) void qnade_sample_kernel(
    const float* __restrict__ W1, const float* __restrict__ b1,
    const float* __restrict__ W2, const float* __restrict__ b2,
    float* __restrict__ wav_out, float* __restrict__ x_out) {
  __shared__ float sW1T[D_IN][H_DIM];  // W1 transposed: column d contiguous
  __shared__ float sW2[2][H_DIM];
  __shared__ float sB1[H_DIM];
  __shared__ float sB2[2];
  __shared__ float sX[SPB][D_IN];

  const int tid = threadIdx.x;
  for (int idx = tid; idx < H_DIM * D_IN; idx += 256) {
    const int i = idx / D_IN, d = idx % D_IN;
    sW1T[d][i] = W1[idx];
  }
  for (int idx = tid; idx < 2 * H_DIM; idx += 256)
    sW2[idx / H_DIM][idx % H_DIM] = W2[idx];
  for (int idx = tid; idx < H_DIM; idx += 256) sB1[idx] = b1[idx];
  if (tid < 2) sB2[tid] = b2[tid];
  __syncthreads();

  const int wave      = tid >> 5;
  const unsigned lane = lane_id();
  const int s         = blockIdx.x * SPB + wave;

  float acc[8];  // this lane's 8 strided slots of the 256-wide pre-activation
#pragma unroll
  for (int r = 0; r < 8; r++) acc[r] = 0.0f;
  float wav = 1.0f;

  for (int d = 0; d < D_IN; d++) {
    float p0 = 0.0f, p1 = 0.0f;
#pragma unroll
    for (int r = 0; r < 8; r++) {
      const int i  = r * 32 + lane;
      const float th = fast_tanh(acc[r] + sB1[i]);
      p0 = fmaf(th, sW2[0][i], p0);
      p1 = fmaf(th, sW2[1][i], p1);
    }
    // wave32 butterfly reduction -> every lane holds the full dot products
#pragma unroll
    for (int off = 16; off > 0; off >>= 1) {
      p0 += __shfl_xor(p0, off, 32);
      p1 += __shfl_xor(p1, off, 32);
    }
    p0 += sB2[0];
    p1 += sB2[1];
    const float inv  = rsqrtf(p0 * p0 + p1 * p1);
    const float psp  = p0 * inv;   // psi_pos
    const float psn  = p1 * inv;   // psi_neg
    const float prob = fminf(psp * psp, 1.0f);

    const uint2 rb = threefry2x32(1u, (unsigned)d, (unsigned)s, 0u);
    const float u  = (float)(rb.x >> 8) * (1.0f / 16777216.0f);
    const float spin = (u < prob) ? 1.0f : -1.0f;
    const float sel  = (spin > 0.0f) ? psp : psn;
    wav *= sel;
    if (lane == 0) sX[wave][d] = spin;

    // incremental prefix update: acc += spin * W1[:, d]
#pragma unroll
    for (int r = 0; r < 8; r++) {
      const int i = r * 32 + lane;
      acc[r] = fmaf(spin, sW1T[d][i], acc[r]);
    }
  }
  if (lane == 0) wav_out[s] = wav;
  __syncthreads();
  // coalesced writeback of the block's 8 sampled rows
  for (int idx = tid; idx < SPB * D_IN; idx += 256) {
    const int w = idx / D_IN, dd = idx % D_IN;
    x_out[(size_t)(blockIdx.x * SPB + w) * D_IN + dd] = sX[w][dd];
  }
}

// ---------------------------------------------------------------------------
// Phase 2: gradients.  8 samples per 256-thread block (8 waves).
//   matmul1: pre1 = xin(64x64) @ W1^T(64x256)  -> h = tanh(pre1 + b1)
//   matmul2: gW1  = g_pre1^T(256x64) @ xin(64x64)   (streamed to HBM)
// Both via v_wmma_f32_16x16x32_f16; xin is exact in fp16.
// ---------------------------------------------------------------------------
__global__ __launch_bounds__(256) void qnade_grad_kernel(
    const float* __restrict__ W1, const float* __restrict__ b1,
    const float* __restrict__ W2, const float* __restrict__ b2,
    const float* __restrict__ x_in,
    float* __restrict__ gW1, float* __restrict__ gb1,
    float* __restrict__ gW2, float* __restrict__ gb2) {
  __shared__ _Float16 sW1h[H_DIM * D_IN];   // fp16 W1, row-major [n][k] (B of matmul1)
  __shared__ float    sB1[H_DIM];
  __shared__ float    sW2[2 * H_DIM];
  __shared__ float    sB2[2];
  __shared__ _Float16 sXA[D_IN * D_IN];     // xin  [d][j]  (A of matmul1)
  __shared__ _Float16 sXT[D_IN * D_IN];     // xin^T [j][d] (B of matmul2)
  __shared__ float    sH[D_IN * H_DIM];     // h [d][i], fp32 (doubles as W1 stage)
  __shared__ _Float16 sGT[H_DIM * D_IN];    // g_pre1^T [i][d] (A of matmul2)
  __shared__ float    sXR[D_IN];
  __shared__ float    sG2[D_IN * 2];        // g_pre2
  __shared__ float    sRed[4][D_IN][2];     // partial sums for output head

  const int tid = threadIdx.x;

  // ---- stage weights once per block ---------------------------------------
#ifdef HAVE_ASYNC_LDS
  // async DMA the 64KB fp32 W1 image into the (idle) sH buffer, then convert
  {
    const char* gsrc = (const char*)W1;
    char* ldst = (char*)sH;
    for (int idx = tid; idx < (H_DIM * D_IN * 4) / 16; idx += 256) {
      __builtin_amdgcn_global_load_async_to_lds_b128(
          (v4i*)(gsrc + idx * 16), (v4i*)(ldst + idx * 16), 0, 0);
    }
#if __has_builtin(__builtin_amdgcn_s_wait_asynccnt)
    __builtin_amdgcn_s_wait_asynccnt(0);
#else
    asm volatile("s_wait_asynccnt 0" ::: "memory");
#endif
    __syncthreads();
    for (int idx = tid; idx < H_DIM * D_IN; idx += 256)
      sW1h[idx] = (_Float16)sH[idx];
  }
#else
  for (int idx = tid; idx < H_DIM * D_IN; idx += 256)
    sW1h[idx] = (_Float16)W1[idx];
#endif
  for (int idx = tid; idx < H_DIM; idx += 256) sB1[idx] = b1[idx];
  for (int idx = tid; idx < 2 * H_DIM; idx += 256) sW2[idx] = W2[idx];
  if (tid < 2) sB2[tid] = b2[tid];

  const int wave      = tid >> 5;
  const unsigned lane = lane_id();

  for (int ss = 0; ss < SPB; ss++) {
    const int s = blockIdx.x * SPB + ss;
    const float* xr = x_in + (size_t)s * D_IN;
    if (ss + 1 < SPB) __builtin_prefetch(xr + D_IN, 0, 3);  // global_prefetch_b8
    __syncthreads();
    if (tid < D_IN) sXR[tid] = xr[tid];
    __syncthreads();

    // build xin (strict-lower-prefix mask of xr) in fp16, plus its transpose
    for (int idx = tid; idx < D_IN * D_IN; idx += 256) {
      const int d = idx >> 6, j = idx & 63;
      const _Float16 v = (j < d) ? (_Float16)sXR[j] : (_Float16)0.0f;
      sXA[d * D_IN + j] = v;
      sXT[j * D_IN + d] = v;
    }
    __syncthreads();

    // ---- matmul1: h = tanh(xin @ W1^T + b1), M=64 N=256 K=64
    // Per wave: ONE A tile (mt = wave&3), 8 B tiles (nt = 2*tt + (wave>>2)).
    {
      const int mt     = wave & 3;
      const int ntbase = wave >> 2;
      const v16h a0 = load_frag_a(sXA + (mt * 16) * D_IN + 0,  D_IN, lane);
      const v16h a1 = load_frag_a(sXA + (mt * 16) * D_IN + 32, D_IN, lane);
      const int grp = lane >> 4;
#pragma unroll
      for (int tt = 0; tt < 8; tt++) {
        const int nt = 2 * tt + ntbase;
        v16h b0 = load_frag_b(sW1h + (nt * 16) * D_IN + 0,  D_IN, lane);
        v16h b1f = load_frag_b(sW1h + (nt * 16) * D_IN + 32, D_IN, lane);
        v8f c = {};
        c = __builtin_amdgcn_wmma_f32_16x16x32_f16(false, a0, false, b0,
                                                   (short)0, c, false, false);
        c = __builtin_amdgcn_wmma_f32_16x16x32_f16(false, a1, false, b1f,
                                                   (short)0, c, false, false);
        const int n = nt * 16 + (lane & 15);
#pragma unroll
        for (int r = 0; r < 8; r++) {
          const int m = mt * 16 + r + grp * 8;
          sH[m * H_DIM + n] = fast_tanh(c[r] + sB1[n]);
        }
      }
    }
    __syncthreads();

    // ---- output head: pre2 = h @ W2^T + b2 (K split 4 ways over threads)
    {
      const int dq = tid & 63, q = tid >> 6;
      float p0 = 0.0f, p1 = 0.0f;
      const int i0 = q * 64;
#pragma unroll 8
      for (int i = i0; i < i0 + 64; i++) {
        const float hv = sH[dq * H_DIM + i];
        p0 = fmaf(hv, sW2[i], p0);
        p1 = fmaf(hv, sW2[H_DIM + i], p1);
      }
      sRed[q][dq][0] = p0;
      sRed[q][dq][1] = p1;
    }
    __syncthreads();
    if (tid < D_IN) {
      const int d = tid;
      const float p0 = sB2[0] + sRed[0][d][0] + sRed[1][d][0] +
                       sRed[2][d][0] + sRed[3][d][0];
      const float p1 = sB2[1] + sRed[0][d][1] + sRed[1][d][1] +
                       sRed[2][d][1] + sRed[3][d][1];
      const float inv = rsqrtf(p0 * p0 + p1 * p1);
      const float o0 = p0 * inv, o1 = p1 * inv;
      const bool  pos = sXR[d] > 0.0f;
      const float sel = pos ? o0 : o1;
      // d/dpre2 of log|out_s| through L2 normalization:
      // g_pre2[k] = ((k==s ? 1/sel : 0) - out[k]) / ||pre2||
      sG2[d * 2 + 0] = ((pos ? 1.0f / sel : 0.0f) - o0) * inv;
      sG2[d * 2 + 1] = ((pos ? 0.0f : 1.0f / sel) - o1) * inv;
    }
    __syncthreads();

    // ---- g_h = g_pre2 @ W2 ; g_pre1 = g_h * (1-h^2) ; gb1/gW2 reductions
    {
      const int i = tid;  // 256 threads == H_DIM columns
      const float w0 = sW2[i], w1 = sW2[H_DIM + i];
      float gb1v = 0.0f, gw2a = 0.0f, gw2b = 0.0f;
      for (int d = 0; d < D_IN; d++) {
        const float hv = sH[d * H_DIM + i];
        const float g0 = sG2[d * 2 + 0], g1 = sG2[d * 2 + 1];
        const float gh = g0 * w0 + g1 * w1;
        const float gp = gh * (1.0f - hv * hv);
        sGT[i * D_IN + d] = (_Float16)gp;
        gb1v += gp;
        gw2a = fmaf(g0, hv, gw2a);
        gw2b = fmaf(g1, hv, gw2b);
      }
      gb1[(size_t)s * H_DIM + i] = gb1v;
      gW2[(size_t)s * 2 * H_DIM + i] = gw2a;
      gW2[(size_t)s * 2 * H_DIM + H_DIM + i] = gw2b;
    }
    if (tid < 2) {
      float a2 = 0.0f;
      for (int d = 0; d < D_IN; d++) a2 += sG2[d * 2 + tid];
      gb2[(size_t)s * 2 + tid] = a2;
    }
    __syncthreads();

    // ---- matmul2: gW1 = g_pre1^T @ xin, M=256 N=64 K=64
    // Per wave: TWO A tiles (rows wave*16 and (wave+8)*16), 4 shared B tiles.
    {
      float* __restrict__ gout = gW1 + (size_t)s * H_DIM * D_IN;
      const int grp = lane >> 4;
      const v16h a00 = load_frag_a(sGT + (wave * 16) * D_IN + 0,  D_IN, lane);
      const v16h a01 = load_frag_a(sGT + (wave * 16) * D_IN + 32, D_IN, lane);
      const v16h a10 = load_frag_a(sGT + ((wave + 8) * 16) * D_IN + 0,  D_IN, lane);
      const v16h a11 = load_frag_a(sGT + ((wave + 8) * 16) * D_IN + 32, D_IN, lane);
#pragma unroll
      for (int nt = 0; nt < 4; nt++) {
        v16h b0 = load_frag_b(sXT + (nt * 16) * D_IN + 0,  D_IN, lane);
        v16h b1f = load_frag_b(sXT + (nt * 16) * D_IN + 32, D_IN, lane);
        v8f c0 = {}, c1 = {};
        c0 = __builtin_amdgcn_wmma_f32_16x16x32_f16(false, a00, false, b0,
                                                    (short)0, c0, false, false);
        c0 = __builtin_amdgcn_wmma_f32_16x16x32_f16(false, a01, false, b1f,
                                                    (short)0, c0, false, false);
        c1 = __builtin_amdgcn_wmma_f32_16x16x32_f16(false, a10, false, b0,
                                                    (short)0, c1, false, false);
        c1 = __builtin_amdgcn_wmma_f32_16x16x32_f16(false, a11, false, b1f,
                                                    (short)0, c1, false, false);
        const int j = nt * 16 + (lane & 15);
#pragma unroll
        for (int r = 0; r < 8; r++) {
          const int i0 = wave * 16 + r + grp * 8;
          const int i1 = (wave + 8) * 16 + r + grp * 8;
          gout[(size_t)i0 * D_IN + j] = c0[r];  // 16 consecutive lanes -> coalesced
          gout[(size_t)i1 * D_IN + j] = c1[r];
        }
      }
    }
    __syncthreads();
  }
}

// ---------------------------------------------------------------------------
extern "C" void kernel_launch(void* const* d_in, const int* in_sizes, int n_in,
                              void* d_out, int out_size, void* d_ws,
                              size_t ws_size, hipStream_t stream) {
  (void)in_sizes; (void)n_in; (void)out_size; (void)d_ws; (void)ws_size;
  const float* W1 = (const float*)d_in[0];
  const float* b1 = (const float*)d_in[1];
  const float* W2 = (const float*)d_in[2];
  const float* b2 = (const float*)d_in[3];
  // d_in[4] = N_samples (fixed 2048 per setup)

  float* out = (float*)d_out;
  float* wav = out;                                   // [2048]
  float* x   = wav + NS;                              // [2048,64]
  float* gW1 = x + (size_t)NS * D_IN;                 // [2048,256,64]
  float* gb1 = gW1 + (size_t)NS * H_DIM * D_IN;       // [2048,256]
  float* gW2 = gb1 + (size_t)NS * H_DIM;              // [2048,2,256]
  float* gb2 = gW2 + (size_t)NS * 2 * H_DIM;          // [2048,2]

  qnade_sample_kernel<<<NS / SPB, 256, 0, stream>>>(W1, b1, W2, b2, wav, x);
  qnade_grad_kernel<<<NS / SPB, 256, 0, stream>>>(W1, b1, W2, b2, x,
                                                  gW1, gb1, gW2, gb2);
}
